// Head_18287970747099
// MI455X (gfx1250) — compile-verified, matrix operand order checked by reading, not run on hardware
//
#include <hip/hip_runtime.h>
#include <hip/hip_bf16.h>

// ---------------------------------------------------------------------------
// CDNA5 (gfx1250) wave32 WMMA attention head.
//   B=8, T=1024, C=512, H=64, BLOCK_SIZE=50 (rel window +-49, 99 rows)
// ---------------------------------------------------------------------------

typedef __attribute__((ext_vector_type(16))) __bf16 v16bf;
typedef __attribute__((ext_vector_type(8)))  __bf16 bf16x8;
typedef __attribute__((ext_vector_type(4)))  __bf16 bf16x4;
typedef __attribute__((ext_vector_type(8)))  float  floatx8;
typedef __attribute__((ext_vector_type(4)))  float  floatx4;

union V16 { v16bf v; bf16x8 h[2]; };

#define BQ 8
#define TT 1024
#define CC 512
#define HH 64
#define RR 112   // 99 rel-pos rows padded to 7*16

static __device__ __forceinline__ floatx8 wmma_bf16(const V16& a, const V16& b, floatx8 c) {
  // D = A(16x32 bf16) x B(32x16 bf16) + C(16x16 f32)
  return __builtin_amdgcn_wmma_f32_16x16x32_bf16(false, a.v, false, b.v, (short)0, c,
                                                 false, false);
}

// ---------------------------------------------------------------------------
// Prep: fp32 -> bf16 conversions / transposes
// ---------------------------------------------------------------------------
__global__ void cvt_x_kernel(const float* __restrict__ x, __bf16* __restrict__ xb, int n4) {
  int i = blockIdx.x * blockDim.x + threadIdx.x;
  if (i >= n4) return;
  const floatx4 f = *(const floatx4*)(x + (size_t)i * 4);
  bf16x4 o;
  o[0] = (__bf16)f[0]; o[1] = (__bf16)f[1]; o[2] = (__bf16)f[2]; o[3] = (__bf16)f[3];
  *(bf16x4*)(xb + (size_t)i * 4) = o;
}

__global__ void prep_w_kernel(const float* __restrict__ Wq, const float* __restrict__ Wk,
                              const float* __restrict__ Wv, __bf16* __restrict__ Wt) {
  // Wt[mat][h][c] = W[mat][c][h]  (bf16, transposed for contiguous B-operand loads)
  int i = blockIdx.x * blockDim.x + threadIdx.x;
  if (i >= 3 * HH * CC) return;
  int mat = i / (HH * CC);
  int rem = i - mat * (HH * CC);
  int h = rem / CC;
  int c = rem - h * CC;
  const float* W = (mat == 0) ? Wq : ((mat == 1) ? Wk : Wv);
  Wt[i] = (__bf16)W[(size_t)c * HH + h];
}

__global__ void prep_e_kernel(const float* __restrict__ E, __bf16* __restrict__ Er) {
  // Er: [112][64] bf16, rows 99..111 zero
  int i = blockIdx.x * blockDim.x + threadIdx.x;
  if (i >= RR * HH) return;
  int r = i / HH;
  Er[i] = (r < 99) ? (__bf16)E[i] : (__bf16)0.0f;
}

// ---------------------------------------------------------------------------
// QKV projection GEMM: [B*T,512]x[512,64] via 16x16x32 bf16 WMMA.
//   grid (512 m-tiles, 3 mats), block 128 (4 waves = 4 n-tiles of 16)
//   Q,K stored row-major bf16 [B*T,64]; V stored transposed bf16 [B][64][T].
// ---------------------------------------------------------------------------
__global__ __launch_bounds__(128) void qkv_kernel(
    const __bf16* __restrict__ xb, const __bf16* __restrict__ Wt,
    __bf16* __restrict__ Qb, __bf16* __restrict__ Kb, __bf16* __restrict__ Vt) {
  const int lane   = threadIdx.x & 31;
  const int wave   = threadIdx.x >> 5;
  const int mtile  = blockIdx.x;
  const int mat    = blockIdx.y;
  const int n0     = wave * 16;
  const int t0     = mtile * 16;
  const int laneLo = lane & 15;
  const int hiK    = (lane >> 4) << 3;   // lanes 0-15: K 0..7/16..23; 16-31: 8..15/24..31
  const int rbase  = (lane >> 4) << 3;   // D rows r / r+8

  const __bf16* arow = xb + (size_t)(t0 + laneLo) * CC + hiK;
  const __bf16* brow = Wt + ((size_t)mat * HH + n0 + laneLo) * CC + hiK;

  floatx8 acc = {};
#pragma unroll 4
  for (int kk = 0; kk < CC; kk += 32) {
    V16 a, b;
    a.h[0] = *(const bf16x8*)(arow + kk);
    a.h[1] = *(const bf16x8*)(arow + kk + 16);
    b.h[0] = *(const bf16x8*)(brow + kk);
    b.h[1] = *(const bf16x8*)(brow + kk + 16);
    acc = wmma_bf16(a, b, acc);
  }

  const int col = n0 + laneLo;
  if (mat < 2) {
    __bf16* dst = (mat == 0) ? Qb : Kb;
#pragma unroll
    for (int r = 0; r < 8; ++r)
      dst[(size_t)(t0 + rbase + r) * HH + col] = (__bf16)acc[r];
  } else {
    const int bq   = mtile >> 6;            // 64 m-tiles per batch
    const int tloc = (t0 & (TT - 1)) + rbase;
    bf16x8 pack;
#pragma unroll
    for (int r = 0; r < 8; ++r) pack[r] = (__bf16)acc[r];
    *(bf16x8*)(Vt + ((size_t)bq * HH + col) * TT + tloc) = pack;  // contiguous in t
  }
}

// ---------------------------------------------------------------------------
// qrel[b,t,r] = q[b,t,:] . rel_pos_emb[r,:]  — [B*T,64]x[64,112] WMMA GEMM
//   grid (512 m-tiles, 7 n-tiles), block 32
// ---------------------------------------------------------------------------
__global__ __launch_bounds__(32) void qrel_kernel(
    const __bf16* __restrict__ Qb, const __bf16* __restrict__ Er,
    float* __restrict__ qrel) {
  const int lane   = threadIdx.x & 31;
  const int t0     = blockIdx.x * 16;
  const int r0     = blockIdx.y * 16;
  const int laneLo = lane & 15;
  const int hiK    = (lane >> 4) << 3;
  const int rbase  = (lane >> 4) << 3;

  const __bf16* arow = Qb + (size_t)(t0 + laneLo) * HH + hiK;
  const __bf16* brow = Er + (size_t)(r0 + laneLo) * HH + hiK;

  floatx8 acc = {};
#pragma unroll
  for (int kk = 0; kk < HH; kk += 32) {
    V16 a, b;
    a.h[0] = *(const bf16x8*)(arow + kk);
    a.h[1] = *(const bf16x8*)(arow + kk + 16);
    b.h[0] = *(const bf16x8*)(brow + kk);
    b.h[1] = *(const bf16x8*)(brow + kk + 16);
    acc = wmma_bf16(a, b, acc);
  }
  const int col = r0 + laneLo;
#pragma unroll
  for (int r = 0; r < 8; ++r)
    qrel[(size_t)(t0 + rbase + r) * RR + col] = acc[r];
}

// ---------------------------------------------------------------------------
// Flash attention: one wave per (b, 16-row query tile); 32-wide causal chunks.
//   grid 128, block 128 (4 waves -> 4 consecutive tiles). LDS: P transpose.
// ---------------------------------------------------------------------------
__global__ __launch_bounds__(128) void attn_kernel(
    const __bf16* __restrict__ Qb, const __bf16* __restrict__ Kb,
    const __bf16* __restrict__ Vt, const float* __restrict__ qrel,
    float* __restrict__ out) {
  __shared__ __bf16 Pl[4][16 * 32];

  const int lane   = threadIdx.x & 31;
  const int wave   = threadIdx.x >> 5;
  const int tile   = blockIdx.x * 4 + wave;   // 0..511
  const int bq     = tile >> 6;
  const int jt     = tile & 63;
  const int t0     = jt * 16;
  const int laneLo = lane & 15;
  const int hiK    = (lane >> 4) << 3;
  const int rbase  = (lane >> 4) << 3;
  const int t_hi   = t0 + 15;

  // Q tile as A-operand for both K-steps (16x64 bf16, lives in VGPRs)
  const __bf16* qrow = Qb + ((size_t)bq * TT + t0 + laneLo) * HH + hiK;
  V16 aq0, aq1;
  aq0.h[0] = *(const bf16x8*)(qrow +  0);
  aq0.h[1] = *(const bf16x8*)(qrow + 16);
  aq1.h[0] = *(const bf16x8*)(qrow + 32);
  aq1.h[1] = *(const bf16x8*)(qrow + 48);

  floatx8 acc[4] = {};                // 16x64 fp32 output accumulator
  float m_i[8], l_i[8];
#pragma unroll
  for (int r = 0; r < 8; ++r) { m_i[r] = -__builtin_inff(); l_i[r] = 0.0f; }

  const float  scale   = 0.125f;      // 64^-0.5
  const float* qr_base = qrel + (size_t)bq * TT * RR;
  __bf16*      plds    = Pl[wave];

  for (int s0 = 0; s0 <= t_hi; s0 += 32) {
    if (s0 + 32 <= t_hi) {            // prefetch next chunk's K/V
      __builtin_prefetch(Kb + ((size_t)bq * TT + s0 + 32 + laneLo) * HH, 0, 0);
      __builtin_prefetch(Vt + ((size_t)bq * HH + laneLo) * TT + s0 + 32, 0, 0);
    }

    // ---- scores: S[16x32] = Q . K^T, two 16x16 D-tiles, K=64 in 2 steps ----
    floatx8 S[2];
#pragma unroll
    for (int st = 0; st < 2; ++st) {
      const int scol = s0 + st * 16 + laneLo;          // B column = key index
      const __bf16* krow = Kb + ((size_t)bq * TT + scol) * HH + hiK;
      V16 b0, b1;
      b0.h[0] = *(const bf16x8*)(krow +  0);
      b0.h[1] = *(const bf16x8*)(krow + 16);
      b1.h[0] = *(const bf16x8*)(krow + 32);
      b1.h[1] = *(const bf16x8*)(krow + 48);
      floatx8 s_ = {};
      s_ = wmma_bf16(aq0, b0, s_);
      s_ = wmma_bf16(aq1, b1, s_);
      S[st] = s_;
    }

    // ---- scale + relative-position bias gather + causal mask ----
#pragma unroll
    for (int st = 0; st < 2; ++st) {
      const int scol = s0 + st * 16 + laneLo;
#pragma unroll
      for (int r = 0; r < 8; ++r) {
        const int t     = t0 + rbase + r;
        int       delta = scol - t;
        delta = (delta < -49) ? -49 : ((delta > 49) ? 49 : delta);
        const float bias = qr_base[(size_t)t * RR + (delta + 49)];
        const float val  = S[st][r] * scale + bias;
        S[st][r] = (scol <= t) ? val : -__builtin_inff();
      }
    }

    // ---- online softmax (rows live across 16-lane halves) ----
    float mc[8];
#pragma unroll
    for (int r = 0; r < 8; ++r) mc[r] = fmaxf(S[0][r], S[1][r]);
#pragma unroll
    for (int off = 8; off >= 1; off >>= 1)
#pragma unroll
      for (int r = 0; r < 8; ++r) mc[r] = fmaxf(mc[r], __shfl_xor(mc[r], off, 32));

    float corr[8], psum[8];
#pragma unroll
    for (int r = 0; r < 8; ++r) {
      const float mnew = fmaxf(m_i[r], mc[r]);
      corr[r] = __expf(m_i[r] - mnew);
      m_i[r]  = mnew;
    }
#pragma unroll
    for (int st = 0; st < 2; ++st)
#pragma unroll
      for (int r = 0; r < 8; ++r) S[st][r] = __expf(S[st][r] - m_i[r]);
#pragma unroll
    for (int r = 0; r < 8; ++r) psum[r] = S[0][r] + S[1][r];
#pragma unroll
    for (int off = 8; off >= 1; off >>= 1)
#pragma unroll
      for (int r = 0; r < 8; ++r) psum[r] += __shfl_xor(psum[r], off, 32);
#pragma unroll
    for (int r = 0; r < 8; ++r) l_i[r] = l_i[r] * corr[r] + psum[r];
#pragma unroll
    for (int dt = 0; dt < 4; ++dt)
#pragma unroll
      for (int r = 0; r < 8; ++r) acc[dt][r] *= corr[r];

    // ---- transpose P (D-layout -> A-layout) through LDS ----
#pragma unroll
    for (int st = 0; st < 2; ++st)
#pragma unroll
      for (int r = 0; r < 8; ++r)
        plds[(rbase + r) * 32 + st * 16 + laneLo] = (__bf16)S[st][r];
    asm volatile("s_wait_dscnt 0x0" ::: "memory");   // LDS RAW fence (single wave)

    V16 ap;
    const __bf16* prow = plds + laneLo * 32 + hiK;
    ap.h[0] = *(const bf16x8*)(prow);
    ap.h[1] = *(const bf16x8*)(prow + 16);

    // ---- acc += P(16x32) . V(32x64), V^T rows are contiguous along s ----
#pragma unroll
    for (int dt = 0; dt < 4; ++dt) {
      const __bf16* vrow = Vt + ((size_t)bq * HH + dt * 16 + laneLo) * TT + s0 + hiK;
      V16 bv;
      bv.h[0] = *(const bf16x8*)(vrow);
      bv.h[1] = *(const bf16x8*)(vrow + 16);
      acc[dt] = wmma_bf16(ap, bv, acc[dt]);
    }
  }

  // ---- epilogue: normalize and store fp32 ----
  float inv_l[8];
#pragma unroll
  for (int r = 0; r < 8; ++r) inv_l[r] = 1.0f / l_i[r];
#pragma unroll
  for (int dt = 0; dt < 4; ++dt)
#pragma unroll
    for (int r = 0; r < 8; ++r)
      out[((size_t)bq * TT + t0 + rbase + r) * HH + dt * 16 + laneLo] =
          acc[dt][r] * inv_l[r];
}

// ---------------------------------------------------------------------------
extern "C" void kernel_launch(void* const* d_in, const int* in_sizes, int n_in,
                              void* d_out, int out_size, void* d_ws, size_t ws_size,
                              hipStream_t stream) {
  const float* x  = (const float*)d_in[0];   // [8,1024,512]
  const float* Wq = (const float*)d_in[1];   // [512,64]
  const float* Wk = (const float*)d_in[2];
  const float* Wv = (const float*)d_in[3];
  const float* E  = (const float*)d_in[4];   // [99,64]
  float* out = (float*)d_out;                // [8,1024,64]

  // workspace layout (bf16 unless noted), ~15.4 MB total
  __bf16* xb = (__bf16*)d_ws;                          // 4,194,304
  __bf16* Wt = xb + (size_t)BQ * TT * CC;              //    98,304
  __bf16* Er = Wt + (size_t)3 * HH * CC;               //     7,168
  __bf16* Qb = Er + (size_t)RR * HH;                   //   524,288
  __bf16* Kb = Qb + (size_t)BQ * TT * HH;              //   524,288
  __bf16* Vt = Kb + (size_t)BQ * TT * HH;              //   524,288
  float* qrel = (float*)(Vt + (size_t)BQ * TT * HH);   // 917,504 fp32 (16B-aligned)

  const int n4 = BQ * TT * CC / 4;
  cvt_x_kernel <<<(n4 + 255) / 256, 256, 0, stream>>>(x, xb, n4);
  prep_w_kernel<<<(3 * HH * CC + 255) / 256, 256, 0, stream>>>(Wq, Wk, Wv, Wt);
  prep_e_kernel<<<(RR * HH + 255) / 256, 256, 0, stream>>>(E, Er);

  qkv_kernel <<<dim3(BQ * TT / 16, 3), 128, 0, stream>>>(xb, Wt, Qb, Kb, Vt);
  qrel_kernel<<<dim3(BQ * TT / 16, RR / 16), 32, 0, stream>>>(Qb, Er, qrel);
  attn_kernel<<<BQ * TT / 16 / 4, 128, 0, stream>>>(Qb, Kb, Vt, qrel, out);
}
